// NonLocalAttention_76201309766491
// MI455X (gfx1250) — matrix operand order repaired and checked by reading, not compiled, not run on hardware
//
#include <hip/hip_runtime.h>
#include <math.h>

typedef __attribute__((ext_vector_type(16))) __bf16 v16bf;
typedef __attribute__((ext_vector_type(8)))  __bf16 v8bf;
typedef __attribute__((ext_vector_type(8)))  float  v8f;
typedef __attribute__((ext_vector_type(4)))  float  v4f;

#define T_    3
#define C_    128
#define H_    96
#define W_    96
#define HW_   (H_*W_)            // 9216
#define NPIX  (T_*HW_)           // 27648
#define NQ    1728               // 3 * 24 * 24
#define ROWS  (NQ*49)            // 84672
#define NSH   81
#define KTOP  10

__device__ __forceinline__ int refl(int i, int n) {
    i = (i < 0) ? -i : i;
    return (i >= n) ? (2 * (n - 1) - i) : i;
}

__device__ __forceinline__ void query_coords(int q, int& qt, int& qy, int& qx) {
    qt = q / 576;
    int rem = q % 576;
    qy = (rem / 24) * 4;
    qx = (rem % 24) * 4;
}

// ---------------------------------------------------------------------------
// prep: vid (T,C,H,W) f32 -> xT (pix, C) bf16 ; W* f32 -> bf16 ; zero acc/zv
// ---------------------------------------------------------------------------
__global__ void prep_kernel(const float* __restrict__ vid,
                            const float* __restrict__ Wq, const float* __restrict__ Wk,
                            const float* __restrict__ Wv, const float* __restrict__ Wp,
                            __bf16* __restrict__ xT,
                            __bf16* __restrict__ Wqb, __bf16* __restrict__ Wkb,
                            __bf16* __restrict__ Wvb, __bf16* __restrict__ Wpb,
                            float* __restrict__ acc, float* __restrict__ zv) {
    int idx = blockIdx.x * 256 + threadIdx.x;           // 0 .. NPIX*C_-1
    if (idx < NPIX * C_) {
        int pix = idx >> 7, c = idx & 127;
        int t = pix / HW_, yx = pix % HW_;
        xT[idx] = (__bf16)vid[((size_t)t * C_ + c) * HW_ + yx];
        acc[idx] = 0.0f;
    }
    if (idx < C_ * C_) {
        Wqb[idx] = (__bf16)Wq[idx];
        Wkb[idx] = (__bf16)Wk[idx];
        Wvb[idx] = (__bf16)Wv[idx];
        Wpb[idx] = (__bf16)Wp[idx];
    }
    if (idx < NPIX) zv[idx] = 0.0f;
}

// ---------------------------------------------------------------------------
// Fused q/k/v conv1x1 GEMM.  One wave per 16-pixel tile; B-tiles (activations)
// preloaded once and reused for all three weight matrices (96 WMMAs / wave).
// xT is (pix, C) bf16 -> each B operand is one contiguous 32B load per lane.
// ---------------------------------------------------------------------------
__global__ void conv_qkv_kernel(const __bf16* __restrict__ xT,
                                const __bf16* __restrict__ Wqb,
                                const __bf16* __restrict__ Wkb,
                                const __bf16* __restrict__ Wvb,
                                const float* __restrict__ bq,
                                const float* __restrict__ bk,
                                const float* __restrict__ bv,
                                __bf16* __restrict__ qT, __bf16* __restrict__ kT,
                                float* __restrict__ vT) {
    const int wave = threadIdx.x >> 5;
    const int lane = threadIdx.x & 31;
    const int tile = blockIdx.x * 8 + wave;
    if (tile >= NPIX / 16) return;
    const int pixbase = tile * 16;
    const int n  = lane & 15;   // column (pixel within tile)
    const int hi = lane >> 4;   // lane half

    // Preload all four K-step B tiles:  B[k,n] = xT[pixbase+n, kb*32 + hi*16 + t]
    v16bf btile[4];
    const __bf16* xrow = xT + (size_t)(pixbase + n) * C_ + hi * 16;
#pragma unroll
    for (int kb = 0; kb < 4; ++kb)
        btile[kb] = *(const v16bf*)(xrow + kb * 32);

    const int m = n;            // A-row (output channel within tile)
    const int khalf = hi * 16;

#pragma unroll
    for (int mat = 0; mat < 3; ++mat) {
        const __bf16* Wb  = (mat == 0) ? Wqb : (mat == 1) ? Wkb : Wvb;
        const float*  bias = (mat == 0) ? bq : (mat == 1) ? bk : bv;
#pragma unroll
        for (int mt = 0; mt < 8; ++mt) {
            v8f acc = {};
#pragma unroll
            for (int kb = 0; kb < 4; ++kb) {
                const __bf16* ar = Wb + (size_t)(mt * 16 + m) * C_ + kb * 32 + khalf;
                union { v16bf v; v8bf h[2]; } au;
                au.h[0] = *(const v8bf*)(ar);
                au.h[1] = *(const v8bf*)(ar + 8);
                acc = __builtin_amdgcn_wmma_f32_16x16x32_bf16(
                        false, au.v, false, btile[kb], (short)0, acc, false, false);
            }
            const int mbase = mt * 16 + hi * 8;           // 8 consecutive rows / lane
            const size_t obase = (size_t)(pixbase + n) * C_ + mbase;
            if (mat < 2) {
                v8bf o;
#pragma unroll
                for (int r = 0; r < 8; ++r) o[r] = (__bf16)(acc[r] + bias[mbase + r]);
                *(v8bf*)((mat == 0 ? qT : kT) + obase) = o;
            } else {
                float* dst = vT + obase;
#pragma unroll
                for (int r = 0; r < 8; ++r) dst[r] = acc[r] + bias[mbase + r];
            }
        }
    }
}

// ---------------------------------------------------------------------------
// Per-(head, query) wave: 81 shifted patch correlations, top-10, softmax.
// ---------------------------------------------------------------------------
__global__ void dist_topk_kernel(const __bf16* __restrict__ qT,
                                 const __bf16* __restrict__ kT,
                                 float* __restrict__ wOut, int* __restrict__ selOut) {
    __shared__ float sd[8][96];
    const int wave = threadIdx.x >> 5;
    const int lane = threadIdx.x & 31;
    const int item = blockIdx.x * 8 + wave;       // h * NQ + q
    const int h = item / NQ;
    const int q = item % NQ;
    int qt, qy, qx; query_coords(q, qt, qy, qx);

    const int s0 = lane, s1 = lane + 32, s2 = lane + 64;
    const int syv[3] = { s0 / 9 - 4, s1 / 9 - 4, s2 / 9 - 4 };
    const int sxv[3] = { s0 % 9 - 4, s1 % 9 - 4, s2 % 9 - 4 };
    float dist[3] = { 0.f, 0.f, 0.f };

    for (int i = 0; i < 7; ++i) {
        const int yq = refl(qy + i - 3, H_);
        for (int j = 0; j < 7; ++j) {
            const int xq = refl(qx + j - 3, W_);
            const __bf16* qp = qT + (size_t)(qt * HW_ + yq * W_ + xq) * C_ + h * 32;
            float qf[32];
#pragma unroll
            for (int c = 0; c < 32; c += 8) {
                v8bf qv = *(const v8bf*)(qp + c);
#pragma unroll
                for (int e = 0; e < 8; ++e) qf[c + e] = (float)qv[e];
            }
#pragma unroll
            for (int u = 0; u < 3; ++u) {
                const int yk = refl(qy + syv[u] + i - 3, H_);
                const int xk = refl(qx + sxv[u] + j - 3, W_);
                const __bf16* kp = kT + (size_t)(qt * HW_ + yk * W_ + xk) * C_ + h * 32;
                float d = 0.f;
#pragma unroll
                for (int c = 0; c < 32; c += 8) {
                    v8bf kv = *(const v8bf*)(kp + c);
#pragma unroll
                    for (int e = 0; e < 8; ++e) d += qf[c + e] * (float)kv[e];
                }
                dist[u] += d;
            }
        }
    }
    if (s0 < NSH) sd[wave][s0] = dist[0];
    if (s1 < NSH) sd[wave][s1] = dist[1];
    if (s2 < NSH) sd[wave][s2] = dist[2];
    __syncthreads();

    if (lane == 0) {
        float vals[KTOP]; int idxs[KTOP];
        for (int kk = 0; kk < KTOP; ++kk) {
            float best = -3.4e38f; int bi = 0;
            for (int s = 0; s < NSH; ++s) {
                float v = sd[wave][s];
                if (v > best) { best = v; bi = s; }
            }
            vals[kk] = best; idxs[kk] = bi;
            sd[wave][bi] = -3.4e38f;
        }
        const float scale = 0.02525381361380526f;  // 1/sqrt(32*49)
        const float mx = vals[0] * scale;          // vals sorted descending
        float e[KTOP], sum = 0.f;
        for (int kk = 0; kk < KTOP; ++kk) { e[kk] = expf(vals[kk] * scale - mx); sum += e[kk]; }
        const float inv = 1.0f / sum;
        for (int kk = 0; kk < KTOP; ++kk) {
            wOut[(size_t)item * KTOP + kk] = e[kk] * inv;
            selOut[(size_t)item * KTOP + kk] = idxs[kk];
        }
    }
}

// ---------------------------------------------------------------------------
// Weighted gather of v patches over top-10 shifts -> P rows (ROWS, C) bf16.
// ---------------------------------------------------------------------------
__global__ void aggregate_kernel(const float* __restrict__ vT,
                                 const float* __restrict__ wIn,
                                 const int* __restrict__ selIn,
                                 __bf16* __restrict__ Pbf) {
    const int q = blockIdx.x;
    __shared__ float sw[4][KTOP];
    __shared__ int   ssy[4][KTOP], ssx[4][KTOP];
    const int tid = threadIdx.x;
    if (tid < 4 * KTOP) {
        const int h = tid / KTOP, kk = tid % KTOP;
        const int item = h * NQ + q;
        sw[h][kk] = wIn[(size_t)item * KTOP + kk];
        const int s = selIn[(size_t)item * KTOP + kk];
        ssy[h][kk] = s / 9 - 4;
        ssx[h][kk] = s % 9 - 4;
    }
    __syncthreads();
    int qt, qy, qx; query_coords(q, qt, qy, qx);

    for (int it = tid; it < 49 * 16; it += 256) {   // (pos, 8-ch group)
        const int pos = it / 16, g = it % 16;
        const int h = g >> 2, c0 = (g & 3) * 8;
        const int i = pos / 7, j = pos % 7;
        float a[8] = {0.f, 0.f, 0.f, 0.f, 0.f, 0.f, 0.f, 0.f};
#pragma unroll
        for (int kk = 0; kk < KTOP; ++kk) {
            const int y = refl(qy + ssy[h][kk] + i - 3, H_);
            const int x = refl(qx + ssx[h][kk] + j - 3, W_);
            const float* vp = vT + (size_t)(qt * HW_ + y * W_ + x) * C_ + h * 32 + c0;
            const float wgt = sw[h][kk];
            v4f v0 = *(const v4f*)(vp);
            v4f v1 = *(const v4f*)(vp + 4);
#pragma unroll
            for (int e = 0; e < 4; ++e) { a[e] += wgt * v0[e]; a[e + 4] += wgt * v1[e]; }
        }
        v8bf o;
#pragma unroll
        for (int e = 0; e < 8; ++e) o[e] = (__bf16)a[e];
        *(v8bf*)(Pbf + ((size_t)q * 49 + pos) * C_ + h * 32 + c0) = o;
    }
}

// ---------------------------------------------------------------------------
// Projection GEMM (rows x 128 x 128) with fused reflect scatter-add epilogue.
// ---------------------------------------------------------------------------
__global__ void proj_scatter_kernel(const __bf16* __restrict__ Pbf,
                                    const __bf16* __restrict__ Wpb,
                                    const float* __restrict__ bp,
                                    float* __restrict__ accBuf) {
    const int wave = threadIdx.x >> 5;
    const int lane = threadIdx.x & 31;
    const int tile = blockIdx.x * 8 + wave;
    if (tile >= ROWS / 16) return;
    const int rowbase = tile * 16;
    const int n  = lane & 15;
    const int hi = lane >> 4;

    v16bf btile[4];
    const __bf16* prow = Pbf + (size_t)(rowbase + n) * C_ + hi * 16;
#pragma unroll
    for (int kb = 0; kb < 4; ++kb)
        btile[kb] = *(const v16bf*)(prow + kb * 32);

    // scatter target for this lane's row
    const int row = rowbase + n;
    const int q = row / 49, pos = row % 49;
    int qt, qy, qx; query_coords(q, qt, qy, qx);
    const int i = pos / 7, j = pos % 7;
    const int y = refl(qy + i - 3, H_);
    const int x = refl(qx + j - 3, W_);
    float* dst0 = accBuf + (size_t)(qt * HW_ + y * W_ + x) * C_;

    const int m = n;
    const int khalf = hi * 16;
#pragma unroll
    for (int mt = 0; mt < 8; ++mt) {
        v8f acc = {};
#pragma unroll
        for (int kb = 0; kb < 4; ++kb) {
            const __bf16* ar = Wpb + (size_t)(mt * 16 + m) * C_ + kb * 32 + khalf;
            union { v16bf v; v8bf h[2]; } au;
            au.h[0] = *(const v8bf*)(ar);
            au.h[1] = *(const v8bf*)(ar + 8);
            acc = __builtin_amdgcn_wmma_f32_16x16x32_bf16(
                    false, au.v, false, btile[kb], (short)0, acc, false, false);
        }
        const int mbase = mt * 16 + hi * 8;
#pragma unroll
        for (int r = 0; r < 8; ++r)
            atomicAdd(dst0 + mbase + r, acc[r] + bp[mbase + r]);
    }
}

// ---------------------------------------------------------------------------
// Coverage counts, then normalize + transpose to (T, C, H, W).
// ---------------------------------------------------------------------------
__global__ void count_kernel(float* __restrict__ zv) {
    const int row = blockIdx.x * 256 + threadIdx.x;
    if (row >= ROWS) return;
    const int q = row / 49, pos = row % 49;
    int qt, qy, qx; query_coords(q, qt, qy, qx);
    const int i = pos / 7, j = pos % 7;
    const int y = refl(qy + i - 3, H_);
    const int x = refl(qx + j - 3, W_);
    atomicAdd(zv + qt * HW_ + y * W_ + x, 1.0f);
}

__global__ void normalize_kernel(const float* __restrict__ accBuf,
                                 const float* __restrict__ zv,
                                 float* __restrict__ out) {
    const int idx = blockIdx.x * 256 + threadIdx.x;
    if (idx >= NPIX * C_) return;
    const int pix = idx >> 7, c = idx & 127;
    const int t = pix / HW_, yx = pix % HW_;
    out[((size_t)t * C_ + c) * HW_ + yx] = accBuf[idx] / zv[pix];
}

// ---------------------------------------------------------------------------
extern "C" void kernel_launch(void* const* d_in, const int* in_sizes, int n_in,
                              void* d_out, int out_size, void* d_ws, size_t ws_size,
                              hipStream_t stream) {
    const float* vid = (const float*)d_in[0];
    const float* Wq  = (const float*)d_in[1];
    const float* bq  = (const float*)d_in[2];
    const float* Wk  = (const float*)d_in[3];
    const float* bk  = (const float*)d_in[4];
    const float* Wv  = (const float*)d_in[5];
    const float* bv  = (const float*)d_in[6];
    const float* Wp  = (const float*)d_in[7];
    const float* bp  = (const float*)d_in[8];
    float* out = (float*)d_out;

    char* ws = (char*)d_ws;
    size_t off = 0;
    auto alloc = [&](size_t bytes) { char* p = ws + off; off += (bytes + 255) & ~(size_t)255; return p; };

    __bf16* xT  = (__bf16*)alloc((size_t)NPIX * C_ * 2);
    __bf16* qT  = (__bf16*)alloc((size_t)NPIX * C_ * 2);
    __bf16* kT  = (__bf16*)alloc((size_t)NPIX * C_ * 2);
    float*  vT  = (float*) alloc((size_t)NPIX * C_ * 4);
    __bf16* Wqb = (__bf16*)alloc((size_t)C_ * C_ * 2);
    __bf16* Wkb = (__bf16*)alloc((size_t)C_ * C_ * 2);
    __bf16* Wvb = (__bf16*)alloc((size_t)C_ * C_ * 2);
    __bf16* Wpb = (__bf16*)alloc((size_t)C_ * C_ * 2);
    float*  wBuf   = (float*) alloc((size_t)4 * NQ * KTOP * 4);
    int*    selBuf = (int*)   alloc((size_t)4 * NQ * KTOP * 4);
    __bf16* Pbf    = (__bf16*)alloc((size_t)ROWS * C_ * 2);
    float*  accBuf = (float*) alloc((size_t)NPIX * C_ * 4);
    float*  zv     = (float*) alloc((size_t)NPIX * 4);

    prep_kernel<<<(NPIX * C_ + 255) / 256, 256, 0, stream>>>(
        vid, Wq, Wk, Wv, Wp, xT, Wqb, Wkb, Wvb, Wpb, accBuf, zv);

    conv_qkv_kernel<<<(NPIX / 16 + 7) / 8, 256, 0, stream>>>(
        xT, Wqb, Wkb, Wvb, bq, bk, bv, qT, kT, vT);

    dist_topk_kernel<<<(4 * NQ) / 8, 256, 0, stream>>>(qT, kT, wBuf, selBuf);

    aggregate_kernel<<<NQ, 256, 0, stream>>>(vT, wBuf, selBuf, Pbf);

    count_kernel<<<(ROWS + 255) / 256, 256, 0, stream>>>(zv);

    proj_scatter_kernel<<<(ROWS / 16 + 7) / 8, 256, 0, stream>>>(Pbf, Wpb, bp, accBuf);

    normalize_kernel<<<(NPIX * C_ + 255) / 256, 256, 0, stream>>>(accBuf, zv, out);
}
